// SlotPathG_44032004718743
// MI455X (gfx1250) — matrix operand order repaired and checked by reading, not compiled, not run on hardware
//
#include <hip/hip_runtime.h>
#include <hip/hip_bf16.h>
#include <math.h>

#define B_    8
#define T_    4096
#define D_    1024
#define S_    64
#define TOPK_ 8
#define DQ_   256
#define DH_   512
#define DR_   (B_ * T_)   // 32768 token rows

typedef __bf16 bf16;
typedef __attribute__((ext_vector_type(16))) __bf16 bf16x16;
typedef __attribute__((ext_vector_type(8)))  float  f32x8;
typedef __attribute__((ext_vector_type(4)))  float  f32x4;

// ---------------------------------------------------------------------------
// WMMA tile helpers (CDNA5 v_wmma_f32_16x16x32_bf16, wave32)
// A 16x32 bf16: lane L -> row M=L%16, K=(L/16)*16 .. +15 contiguous
// B 32x16 bf16 (packed): lane L -> col N=L%16, K=(L/16)*16 .. +15 contiguous
// C/D 16x16 f32: VGPR r -> M=(L/16)*8+r, N=L%16
// ---------------------------------------------------------------------------

__device__ inline f32x8 wmma_bf16(bf16x16 a, bf16x16 b, f32x8 c) {
  return __builtin_amdgcn_wmma_f32_16x16x32_bf16(false, a, false, b,
                                                 (short)0, c, false, false);
}

__device__ inline bf16x16 load_a_f32(const float* base, int ld) {
  const int lane = threadIdx.x & 31;
  const int m  = lane & 15;
  const int kc = (lane >> 4) << 4;
  const float* p = base + (size_t)m * ld + kc;
  bf16x16 a;
#pragma unroll
  for (int j = 0; j < 8; ++j) a[j] = (bf16)p[j];
#pragma unroll
  for (int j = 0; j < 8; ++j) a[8 + j] = (bf16)p[8 + j];
  return a;
}

__device__ inline bf16x16 load_a_bf16(const bf16* base, int ld) {
  const int lane = threadIdx.x & 31;
  const int m  = lane & 15;
  const int kc = (lane >> 4) << 4;
  const bf16* p = base + (size_t)m * ld + kc;
  bf16x16 a;
#pragma unroll
  for (int j = 0; j < 8; ++j) a[j] = p[j];
#pragma unroll
  for (int j = 0; j < 8; ++j) a[8 + j] = p[8 + j];
  return a;
}

__device__ inline bf16x16 load_a(const float* p, int ld) { return load_a_f32(p, ld); }
__device__ inline bf16x16 load_a(const bf16* p, int ld)  { return load_a_bf16(p, ld); }

// packed weight tile: 512 bf16 laid out lane-major, 32 contiguous bytes / lane
__device__ inline bf16x16 load_b_packed(const bf16* tile) {
  const int lane = threadIdx.x & 31;
  return *(const bf16x16*)(tile + lane * 16);
}

// B operand from row-major (N,K) f32 storage (i.e. B^T in memory)
__device__ inline bf16x16 load_b_t_f32(const float* base, int ld) {
  const int lane = threadIdx.x & 31;
  const int n  = lane & 15;
  const int kk = (lane >> 4) << 4;
  const float* p = base + (size_t)n * ld + kk;
  bf16x16 b;
#pragma unroll
  for (int j = 0; j < 16; ++j) b[j] = (bf16)p[j];
  return b;
}

// B operand from row-major (K,N) f32 storage (strided per-lane reads)
__device__ inline bf16x16 load_b_strided_f32(const float* base, int ld) {
  const int lane = threadIdx.x & 31;
  const int n  = lane & 15;
  const int kk = (lane >> 4) << 4;
  const float* p = base + (size_t)kk * ld + n;
  bf16x16 b;
#pragma unroll
  for (int j = 0; j < 16; ++j) b[j] = (bf16)p[(size_t)j * ld];
  return b;
}

__device__ inline float gelu_exact(float v) {
  return 0.5f * v * (1.0f + erff(v * 0.70710678118654752f));
}

// ---------------------------------------------------------------------------
// Weight pack: f32 (K,N) [or (N,K) if transposed] -> WMMA-native bf16 tiles.
// tile index = nt*(K/32)+kt, tile holds 512 bf16, lane-major.
// ---------------------------------------------------------------------------
__global__ void pack_b_kernel(const float* __restrict__ W, bf16* __restrict__ out,
                              int K, int N, int transposed) {
  const size_t total = (size_t)K * N;
  const int ktiles = K >> 5;
  for (size_t i = (size_t)blockIdx.x * blockDim.x + threadIdx.x; i < total;
       i += (size_t)gridDim.x * blockDim.x) {
    const int tile = (int)(i >> 9);
    const int lane = (int)((i >> 4) & 31);
    const int j    = (int)(i & 15);
    const int kt = tile % ktiles;
    const int nt = tile / ktiles;
    const int n = nt * 16 + (lane & 15);
    const int k = kt * 32 + ((lane >> 4) << 4) + j;
    const float v = transposed ? W[(size_t)n * K + k] : W[(size_t)k * N + n];
    out[i] = (bf16)v;
  }
}

// ---------------------------------------------------------------------------
// Prep: slot_mean, fused routing bias c1 = slot_mean @ w1[D:] + b1, zero denom
// ---------------------------------------------------------------------------
__global__ void prep_kernel(const float* __restrict__ slot_init,
                            const float* __restrict__ w1,
                            const float* __restrict__ b1,
                            float* __restrict__ slot_mean,
                            float* __restrict__ c1,
                            float* __restrict__ denom) {
  const int t = threadIdx.x;
  for (int d = t; d < D_; d += 256) {
    float s = 0.f;
    for (int i = 0; i < S_; ++i) s += slot_init[(size_t)i * D_ + d];
    slot_mean[d] = s * (1.0f / S_);
  }
  __syncthreads();
  for (int j = t; j < DH_; j += 256) {
    float s = b1[j];
    for (int d = 0; d < D_; ++d)
      s += slot_mean[d] * w1[(size_t)(D_ + d) * DH_ + j];
    c1[j] = s;
  }
  for (int i = t; i < B_ * S_; i += 256) denom[i] = 0.f;
}

// ---------------------------------------------------------------------------
// Generic tiled GEMM, software-pipelined.
// Block = 4 waves. Each wave owns MW 16-row M tiles and the block's 64-wide
// N group (4 WMMA tiles) -> wave computes (16*MW) x 64 of C.
// k-step kt+1 operands are fetched before issuing kt's WMMAs so the
// s_wait_loadcnt in front of each WMMA covers a full k-step of latency.
// ---------------------------------------------------------------------------
template <typename AT, typename OT, int ACT, int MW>
__global__ __launch_bounds__(128)
void gemm_bf16_kernel(const AT* __restrict__ A, int lda,
                      const bf16* __restrict__ Bp,
                      OT* __restrict__ C, int ldc,
                      const float* __restrict__ bias,
                      int M, int N, int K) {
  const int wave = threadIdx.x >> 5;
  const int lane = threadIdx.x & 31;
  const int m0 = blockIdx.x * (64 * MW) + wave * (16 * MW);
  const int n0 = blockIdx.y * 64;
  const int ktiles = K >> 5;
  f32x8 acc[MW][4] = {};
  bf16x16 a_cur[MW], b_cur[4];
#pragma unroll
  for (int mw = 0; mw < MW; ++mw)
    a_cur[mw] = load_a(A + (size_t)(m0 + 16 * mw) * lda, lda);
#pragma unroll
  for (int nt = 0; nt < 4; ++nt)
    b_cur[nt] = load_b_packed(Bp + (size_t)((n0 >> 4) + nt) * ktiles * 512);

  for (int kt = 0; kt < ktiles; ++kt) {
    const int kn = (kt + 1 < ktiles) ? kt + 1 : kt;  // last prefetch harmless
    bf16x16 a_nxt[MW], b_nxt[4];
#pragma unroll
    for (int mw = 0; mw < MW; ++mw)
      a_nxt[mw] = load_a(A + (size_t)(m0 + 16 * mw) * lda + kn * 32, lda);
#pragma unroll
    for (int nt = 0; nt < 4; ++nt)
      b_nxt[nt] = load_b_packed(
          Bp + ((size_t)((n0 >> 4) + nt) * ktiles + kn) * 512);
#pragma unroll
    for (int mw = 0; mw < MW; ++mw)
#pragma unroll
      for (int nt = 0; nt < 4; ++nt)
        acc[mw][nt] = wmma_bf16(a_cur[mw], b_cur[nt], acc[mw][nt]);
#pragma unroll
    for (int mw = 0; mw < MW; ++mw) a_cur[mw] = a_nxt[mw];
#pragma unroll
    for (int nt = 0; nt < 4; ++nt) b_cur[nt] = b_nxt[nt];
  }

  const int n  = lane & 15;
  const int mb = (lane >> 4) << 3;
#pragma unroll
  for (int mw = 0; mw < MW; ++mw)
#pragma unroll
    for (int nt = 0; nt < 4; ++nt) {
      const int col = n0 + nt * 16 + n;
#pragma unroll
      for (int r = 0; r < 8; ++r) {
        float v = acc[mw][nt][r];
        if (bias) v += bias[col];
        if (ACT == 1) v = gelu_exact(v);
        C[(size_t)(m0 + 16 * mw + mb + r) * ldc + col] = (OT)v;
      }
    }
}

// ---------------------------------------------------------------------------
// Routing: logits = (h @ w2 + b2) / (|tau|+0.1), fused top-8 + softmax,
// scatter into transposed dense alpha (B,S,T) + sparse (idx, w) lists + denom.
// ---------------------------------------------------------------------------
__global__ __launch_bounds__(128)
void route_kernel(const bf16* __restrict__ hb, const bf16* __restrict__ w2p,
                  const float* __restrict__ b2, const float* __restrict__ tau,
                  bf16* __restrict__ alphaT, int* __restrict__ tidx,
                  float* __restrict__ tw, float* __restrict__ denom) {
  __shared__ float ls[4][16][64];
  const int wave = threadIdx.x >> 5;
  const int lane = threadIdx.x & 31;
  const int row0 = blockIdx.x * 64 + wave * 16;
  const int ktiles = DH_ >> 5;  // 16
  f32x8 acc[4] = {};
  for (int kt = 0; kt < ktiles; ++kt) {
    bf16x16 a = load_a_bf16(hb + (size_t)row0 * DH_ + kt * 32, DH_);
#pragma unroll
    for (int nt = 0; nt < 4; ++nt) {
      const bf16* tile = w2p + ((size_t)nt * ktiles + kt) * 512;
      acc[nt] = wmma_bf16(a, load_b_packed(tile), acc[nt]);
    }
  }
  const float sc = 1.0f / (fabsf(tau[0]) + 0.1f);
  const int n  = lane & 15;
  const int mb = (lane >> 4) << 3;
#pragma unroll
  for (int nt = 0; nt < 4; ++nt)
#pragma unroll
    for (int r = 0; r < 8; ++r)
      ls[wave][mb + r][nt * 16 + n] = (acc[nt][r] + b2[nt * 16 + n]) * sc;
  __syncthreads();

  if (lane < 16) {
    const int row = row0 + lane;
    const int b = row >> 12;     // / T_
    const int t = row & (T_ - 1);
    float* lrow = ls[wave][lane];
    unsigned long long picked = 0ull;
    float vals[TOPK_]; int idxs[TOPK_];
#pragma unroll
    for (int p = 0; p < TOPK_; ++p) {
      float best = -3.4e38f; int bi = 0;
      for (int i = 0; i < S_; ++i) {
        const float v = lrow[i];
        if (!((picked >> i) & 1ull) && v > best) { best = v; bi = i; }
      }
      picked |= 1ull << bi;
      vals[p] = best; idxs[p] = bi;
    }
    const float mx = vals[0];
    float w[TOPK_]; float ssum = 0.f;
#pragma unroll
    for (int p = 0; p < TOPK_; ++p) { w[p] = expf(vals[p] - mx); ssum += w[p]; }
    const float inv = 1.0f / ssum;
    bf16* at = alphaT + (size_t)b * S_ * T_ + t;
    for (int i = 0; i < S_; ++i) at[(size_t)i * T_] = (bf16)0.f;
#pragma unroll
    for (int p = 0; p < TOPK_; ++p) {
      const float wp = w[p] * inv;
      at[(size_t)idxs[p] * T_] = (bf16)wp;
      tidx[(size_t)row * TOPK_ + p] = idxs[p];
      tw[(size_t)row * TOPK_ + p] = wp;
      atomicAdd(&denom[b * S_ + idxs[p]], wp);
    }
  }
}

// ---------------------------------------------------------------------------
// slot_input: si[b,s,:] = (alphaT[b] @ x[b]) / (denom+1e-8)
// x tile staged through LDS with f32->bf16 transpose (B operand).
// ---------------------------------------------------------------------------
__global__ __launch_bounds__(128)
void slot_input_kernel(const bf16* __restrict__ alphaT, const float* __restrict__ x,
                       const float* __restrict__ denom, float* __restrict__ si) {
  __shared__ bf16 xs[64][32];  // [n][k]
  const int b  = blockIdx.x;
  const int n0 = blockIdx.y * 64;
  const int wave = threadIdx.x >> 5;
  const int lane = threadIdx.x & 31;
  const bf16*  aT = alphaT + (size_t)b * S_ * T_;
  const float* xb = x + (size_t)b * T_ * D_;
  f32x8 acc[4] = {};
  for (int k0 = 0; k0 < T_; k0 += 32) {
#pragma unroll
    for (int j = 0; j < 16; ++j) {
      const int lin = threadIdx.x + j * 128;
      const int kk = lin >> 6;
      const int nn = lin & 63;
      xs[nn][kk] = (bf16)xb[(size_t)(k0 + kk) * D_ + n0 + nn];
    }
    __syncthreads();
    bf16x16 btile;
    {
      const bf16* p = &xs[wave * 16 + (lane & 15)][(lane >> 4) << 4];
#pragma unroll
      for (int j = 0; j < 16; ++j) btile[j] = p[j];
    }
#pragma unroll
    for (int mt = 0; mt < 4; ++mt) {
      bf16x16 a = load_a_bf16(aT + (size_t)(mt * 16) * T_ + k0, T_);
      acc[mt] = wmma_bf16(a, btile, acc[mt]);
    }
    __syncthreads();
  }
  const int n  = lane & 15;
  const int mb = (lane >> 4) << 3;
  const int col = n0 + wave * 16 + n;
#pragma unroll
  for (int mt = 0; mt < 4; ++mt)
#pragma unroll
    for (int r = 0; r < 8; ++r) {
      const int s = mt * 16 + mb + r;
      si[((size_t)b * S_ + s) * D_ + col] =
          acc[mt][r] / (denom[b * S_ + s] + 1e-8f);
    }
}

// ---------------------------------------------------------------------------
// GRU gate combine: S_new = (1-z)*n + z*slot_init
// ---------------------------------------------------------------------------
__global__ void gru_kernel(const float* __restrict__ gi, const float* __restrict__ gh0,
                           const float* __restrict__ slot_init, float* __restrict__ snew) {
  const int row = blockIdx.x;        // b*64+s
  const int s = row & (S_ - 1);
  const float* gir = gi  + (size_t)row * 3 * D_;
  const float* ghr = gh0 + (size_t)s   * 3 * D_;
  for (int d = threadIdx.x; d < D_; d += 256) {
    const float r = 1.0f / (1.0f + expf(-(gir[d] + ghr[d])));
    const float z = 1.0f / (1.0f + expf(-(gir[D_ + d] + ghr[D_ + d])));
    const float nn = tanhf(gir[2 * D_ + d] + r * ghr[2 * D_ + d]);
    const float s0 = slot_init[(size_t)s * D_ + d];
    snew[(size_t)row * D_ + d] = (1.0f - z) * nn + z * s0;
  }
}

// ---------------------------------------------------------------------------
// Slot attention scores: q @ k^T * DQ^-0.5  (per batch, 64x64)
// ---------------------------------------------------------------------------
__global__ __launch_bounds__(128)
void scores_kernel(const float* __restrict__ q, const float* __restrict__ k,
                   float* __restrict__ scores) {
  const int b = blockIdx.x;
  const int wave = threadIdx.x >> 5;
  const int lane = threadIdx.x & 31;
  const int m0 = wave * 16;
  const float* qb = q + (size_t)b * S_ * DQ_;
  const float* kb = k + (size_t)b * S_ * DQ_;
  f32x8 acc[4] = {};
  for (int kt = 0; kt < (DQ_ >> 5); ++kt) {
    bf16x16 a = load_a_f32(qb + (size_t)m0 * DQ_ + kt * 32, DQ_);
#pragma unroll
    for (int nt = 0; nt < 4; ++nt)
      acc[nt] = wmma_bf16(a, load_b_t_f32(kb + (size_t)(nt * 16) * DQ_ + kt * 32, DQ_),
                          acc[nt]);
  }
  const float sc = 0.0625f;  // 256^-0.5
  const int n  = lane & 15;
  const int mb = (lane >> 4) << 3;
#pragma unroll
  for (int nt = 0; nt < 4; ++nt)
#pragma unroll
    for (int r = 0; r < 8; ++r)
      scores[((size_t)b * S_ + m0 + mb + r) * S_ + nt * 16 + n] = acc[nt][r] * sc;
}

__global__ void attn_softmax_kernel(const float* __restrict__ scores,
                                    float* __restrict__ attn) {
  const int row = blockIdx.x * blockDim.x + threadIdx.x;
  if (row >= B_ * S_) return;
  const float* sr = scores + (size_t)row * S_;
  float mx = -3.4e38f;
  for (int i = 0; i < S_; ++i) mx = fmaxf(mx, sr[i]);
  float ssum = 0.f;
  for (int i = 0; i < S_; ++i) {
    const float e = expf(sr[i] - mx);
    attn[(size_t)row * S_ + i] = e;
    ssum += e;
  }
  const float inv = 1.0f / ssum;
  for (int i = 0; i < S_; ++i) attn[(size_t)row * S_ + i] *= inv;
}

// ---------------------------------------------------------------------------
// y_pre = S_new + attn @ v   (per batch, K=64)
// ---------------------------------------------------------------------------
__global__ __launch_bounds__(128)
void scomm_kernel(const float* __restrict__ attn, const float* __restrict__ v,
                  const float* __restrict__ snew, float* __restrict__ ypre) {
  const int b = blockIdx.x;
  const int wave = threadIdx.x >> 5;
  const int lane = threadIdx.x & 31;
  const int n0 = blockIdx.y * 64 + wave * 16;
  const float* ab = attn + (size_t)b * S_ * S_;
  const float* vb = v + (size_t)b * S_ * D_;
  f32x8 acc[4] = {};
  for (int kt = 0; kt < 2; ++kt) {
    bf16x16 bt = load_b_strided_f32(vb + (size_t)(kt * 32) * D_ + n0, D_);
#pragma unroll
    for (int mt = 0; mt < 4; ++mt) {
      bf16x16 a = load_a_f32(ab + (size_t)(mt * 16) * S_ + kt * 32, S_);
      acc[mt] = wmma_bf16(a, bt, acc[mt]);
    }
  }
  const int n  = lane & 15;
  const int mb = (lane >> 4) << 3;
#pragma unroll
  for (int mt = 0; mt < 4; ++mt)
#pragma unroll
    for (int r = 0; r < 8; ++r) {
      const int s = mt * 16 + mb + r;
      const size_t o = ((size_t)b * S_ + s) * D_ + n0 + n;
      ypre[o] = snew[o] + acc[mt][r];
    }
}

// ---------------------------------------------------------------------------
// LayerNorm over D per slot row
// ---------------------------------------------------------------------------
__global__ void ln_kernel(const float* __restrict__ ypre, const float* __restrict__ g,
                          const float* __restrict__ bta, float* __restrict__ y) {
  const int row = blockIdx.x;
  __shared__ float r1[256], r2[256];
  float s = 0.f, s2 = 0.f;
  for (int d = threadIdx.x; d < D_; d += 256) {
    const float v = ypre[(size_t)row * D_ + d];
    s += v; s2 += v * v;
  }
  r1[threadIdx.x] = s; r2[threadIdx.x] = s2;
  __syncthreads();
  for (int o = 128; o > 0; o >>= 1) {
    if (threadIdx.x < o) { r1[threadIdx.x] += r1[threadIdx.x + o];
                           r2[threadIdx.x] += r2[threadIdx.x + o]; }
    __syncthreads();
  }
  const float mu = r1[0] * (1.0f / D_);
  const float var = r2[0] * (1.0f / D_) - mu * mu;
  const float inv = rsqrtf(var + 1e-5f);
  for (int d = threadIdx.x; d < D_; d += 256)
    y[(size_t)row * D_ + d] =
        (ypre[(size_t)row * D_ + d] - mu) * inv * g[d] + bta[d];
}

// ---------------------------------------------------------------------------
// Output gather: out[b,t,:] = sum_p w_p * sv_wo[b, idx_p, :] + bo
// sv_wo[b] is 256 KB and stays L2-resident (192 MB L2).
// ---------------------------------------------------------------------------
__global__ __launch_bounds__(256)
void out_kernel(const int* __restrict__ tidx, const float* __restrict__ tw,
                const float* __restrict__ svwo, const float* __restrict__ bo,
                float* __restrict__ out) {
  const int row = blockIdx.x;
  const int b = row >> 12;
  __shared__ int   sidx[TOPK_];
  __shared__ float swt[TOPK_];
  if (threadIdx.x < TOPK_) {
    sidx[threadIdx.x] = tidx[(size_t)row * TOPK_ + threadIdx.x];
    swt[threadIdx.x]  = tw[(size_t)row * TOPK_ + threadIdx.x];
  }
  __syncthreads();
  const float* base = svwo + (size_t)b * S_ * D_;
  const int d = threadIdx.x * 4;
  f32x4 acc = *(const f32x4*)(bo + d);
#pragma unroll
  for (int p = 0; p < TOPK_; ++p) {
    const f32x4 vv = *(const f32x4*)(base + (size_t)sidx[p] * D_ + d);
    acc += swt[p] * vv;
  }
  *(f32x4*)(out + (size_t)row * D_ + d) = acc;
}

// ---------------------------------------------------------------------------
// Host launcher
// ---------------------------------------------------------------------------
static inline size_t aln(size_t x) { return (x + 255) & ~(size_t)255; }

extern "C" void kernel_launch(void* const* d_in, const int* in_sizes, int n_in,
                              void* d_out, int out_size, void* d_ws, size_t ws_size,
                              hipStream_t stream) {
  (void)in_sizes; (void)n_in; (void)out_size; (void)ws_size;
  const float* x         = (const float*)d_in[0];
  const float* slot_init = (const float*)d_in[1];
  const float* w1   = (const float*)d_in[2];
  const float* b1   = (const float*)d_in[3];
  const float* w2   = (const float*)d_in[4];
  const float* b2   = (const float*)d_in[5];
  const float* tau  = (const float*)d_in[6];
  const float* Wih  = (const float*)d_in[7];
  const float* Whh  = (const float*)d_in[8];
  const float* bih  = (const float*)d_in[9];
  const float* bhh  = (const float*)d_in[10];
  const float* wq   = (const float*)d_in[11];
  const float* bq   = (const float*)d_in[12];
  const float* wk   = (const float*)d_in[13];
  const float* bk   = (const float*)d_in[14];
  const float* wv   = (const float*)d_in[15];
  const float* bv   = (const float*)d_in[16];
  const float* ln_g = (const float*)d_in[17];
  const float* ln_b = (const float*)d_in[18];
  const float* wvp  = (const float*)d_in[19];
  const float* bvp  = (const float*)d_in[20];
  const float* wo   = (const float*)d_in[21];
  const float* bo   = (const float*)d_in[22];
  float* out = (float*)d_out;

  char* ws = (char*)d_ws;
  size_t off = 0;
  auto alloc = [&](size_t bytes) -> char* {
    char* p = ws + off; off = aln(off + bytes); return p;
  };
  float* slot_mean = (float*)alloc(D_ * 4);
  float* c1        = (float*)alloc(DH_ * 4);
  float* denom     = (float*)alloc(B_ * S_ * 4);
  bf16*  hb        = (bf16*) alloc((size_t)DR_ * DH_ * 2);
  bf16*  alphaT    = (bf16*) alloc((size_t)B_ * S_ * T_ * 2);
  int*   tidx      = (int*)  alloc((size_t)DR_ * TOPK_ * 4);
  float* tw        = (float*)alloc((size_t)DR_ * TOPK_ * 4);
  float* si        = (float*)alloc((size_t)B_ * S_ * D_ * 4);
  float* gh0       = (float*)alloc((size_t)S_ * 3 * D_ * 4);
  float* gi        = (float*)alloc((size_t)B_ * S_ * 3 * D_ * 4);
  float* snew      = (float*)alloc((size_t)B_ * S_ * D_ * 4);
  float* qb        = (float*)alloc((size_t)B_ * S_ * DQ_ * 4);
  float* kb        = (float*)alloc((size_t)B_ * S_ * DQ_ * 4);
  float* vb        = (float*)alloc((size_t)B_ * S_ * D_ * 4);
  float* scores    = (float*)alloc((size_t)B_ * S_ * S_ * 4);
  float* attn      = (float*)alloc((size_t)B_ * S_ * S_ * 4);
  float* ypre      = (float*)alloc((size_t)B_ * S_ * D_ * 4);
  float* yn        = (float*)alloc((size_t)B_ * S_ * D_ * 4);
  float* sv        = (float*)alloc((size_t)B_ * S_ * D_ * 4);
  float* svwo      = (float*)alloc((size_t)B_ * S_ * D_ * 4);
  bf16* w1a_p  = (bf16*)alloc((size_t)D_ * DH_ * 2);
  bf16* w2_p   = (bf16*)alloc((size_t)DH_ * S_ * 2);
  bf16* wihT_p = (bf16*)alloc((size_t)D_ * 3 * D_ * 2);
  bf16* whhT_p = (bf16*)alloc((size_t)D_ * 3 * D_ * 2);
  bf16* wq_p   = (bf16*)alloc((size_t)D_ * DQ_ * 2);
  bf16* wk_p   = (bf16*)alloc((size_t)D_ * DQ_ * 2);
  bf16* wv_p   = (bf16*)alloc((size_t)D_ * D_ * 2);
  bf16* wvp_p  = (bf16*)alloc((size_t)D_ * D_ * 2);
  bf16* wo_p   = (bf16*)alloc((size_t)D_ * D_ * 2);

  // prep + weight packing
  prep_kernel<<<1, 256, 0, stream>>>(slot_init, w1, b1, slot_mean, c1, denom);
  auto pack = [&](const float* W, bf16* P, int K, int N, int tr) {
    const int blocks = (int)(((size_t)K * N + 255) / 256);
    pack_b_kernel<<<blocks, 256, 0, stream>>>(W, P, K, N, tr);
  };
  pack(w1,  w1a_p,  D_,  DH_,     0);   // first D rows of w1
  pack(w2,  w2_p,   DH_, S_,      0);
  pack(Wih, wihT_p, D_,  3 * D_,  1);   // Wih.T
  pack(Whh, whhT_p, D_,  3 * D_,  1);   // Whh.T
  pack(wq,  wq_p,   D_,  DQ_,     0);
  pack(wk,  wk_p,   D_,  DQ_,     0);
  pack(wv,  wv_p,   D_,  D_,      0);
  pack(wvp, wvp_p,  D_,  D_,      0);
  pack(wo,  wo_p,   D_,  D_,      0);

  // 1) h = gelu(x @ w1a + c1)  (M=32768, K=1024, N=512) -> bf16, 32x64/wave
  gemm_bf16_kernel<float, bf16, 1, 2><<<dim3(DR_ / 128, DH_ / 64), 128, 0, stream>>>(
      x, D_, w1a_p, hb, DH_, c1, DR_, DH_, D_);

  // 2) routing logits + top-8 + softmax + alpha scatter
  route_kernel<<<DR_ / 64, 128, 0, stream>>>(hb, w2_p, b2, tau, alphaT, tidx, tw,
                                             denom);

  // 3) si = (alphaT @ x) / denom
  slot_input_kernel<<<dim3(B_, D_ / 64), 128, 0, stream>>>(alphaT, x, denom, si);

  // 4) GRU: gh0 (batch-independent, M=64 -> MW=1) and gi, then gate combine
  gemm_bf16_kernel<float, float, 0, 1><<<dim3(1, 3 * D_ / 64), 128, 0, stream>>>(
      slot_init, D_, whhT_p, gh0, 3 * D_, bhh, S_, 3 * D_, D_);
  gemm_bf16_kernel<float, float, 0, 2><<<dim3(B_ * S_ / 128, 3 * D_ / 64), 128, 0, stream>>>(
      si, D_, wihT_p, gi, 3 * D_, bih, B_ * S_, 3 * D_, D_);
  gru_kernel<<<B_ * S_, 256, 0, stream>>>(gi, gh0, slot_init, snew);

  // 5) slot attention
  gemm_bf16_kernel<float, float, 0, 2><<<dim3(B_ * S_ / 128, DQ_ / 64), 128, 0, stream>>>(
      snew, D_, wq_p, qb, DQ_, bq, B_ * S_, DQ_, D_);
  gemm_bf16_kernel<float, float, 0, 2><<<dim3(B_ * S_ / 128, DQ_ / 64), 128, 0, stream>>>(
      snew, D_, wk_p, kb, DQ_, bk, B_ * S_, DQ_, D_);
  gemm_bf16_kernel<float, float, 0, 2><<<dim3(B_ * S_ / 128, D_ / 64), 128, 0, stream>>>(
      snew, D_, wv_p, vb, D_, bv, B_ * S_, D_, D_);
  scores_kernel<<<B_, 128, 0, stream>>>(qb, kb, scores);
  attn_softmax_kernel<<<(B_ * S_ + 255) / 256, 256, 0, stream>>>(scores, attn);
  scomm_kernel<<<dim3(B_, D_ / 64), 128, 0, stream>>>(attn, vb, snew, ypre);
  ln_kernel<<<B_ * S_, 256, 0, stream>>>(ypre, ln_g, ln_b, yn);

  // 6) slot_values @ wo folded slot-side: sv = y@wvp + bvp ; svwo = sv@wo
  gemm_bf16_kernel<float, float, 0, 2><<<dim3(B_ * S_ / 128, D_ / 64), 128, 0, stream>>>(
      yn, D_, wvp_p, sv, D_, bvp, B_ * S_, D_, D_);
  gemm_bf16_kernel<float, float, 0, 2><<<dim3(B_ * S_ / 128, D_ / 64), 128, 0, stream>>>(
      sv, D_, wo_p, svwo, D_, nullptr, B_ * S_, D_, D_);

  // 7) out[b,t,:] = sum_p w_p * svwo[b, idx_p, :] + bo
  out_kernel<<<DR_, 256, 0, stream>>>(tidx, tw, svwo, bo, out);
}
